// _NeuronGPT2Attention_25409026523689
// MI455X (gfx1250) — matrix operand order repaired and checked by reading, not compiled, not run on hardware
//
#include <hip/hip_runtime.h>
#include <hip/hip_bf16.h>

typedef __bf16 bf16;
typedef __attribute__((ext_vector_type(16))) __bf16 v16bf;
typedef __attribute__((ext_vector_type(8)))  __bf16 v8bf;
typedef __attribute__((ext_vector_type(4)))  __bf16 v4bf;
typedef __attribute__((ext_vector_type(8)))  float  v8f;
typedef int v4i128 __attribute__((vector_size(16)));   // builtin's pointee type

#define WMMA_BF16(a, b, c) \
  __builtin_amdgcn_wmma_f32_16x16x32_bf16(false, (a), false, (b), (short)0, (c), false, false)

#if defined(__has_builtin)
# if __has_builtin(__builtin_amdgcn_global_load_async_to_lds_b128)
#  define HAVE_ASYNC_G2L 1
# endif
#endif

// Async global->LDS 16B copy (ASYNCcnt path); fallback: VGPR round-trip.
__device__ __forceinline__ void g2l_b128(const bf16* g, __bf16* l) {
#ifdef HAVE_ASYNC_G2L
  __builtin_amdgcn_global_load_async_to_lds_b128(
      (__attribute__((address_space(1))) v4i128*)(g),
      (__attribute__((address_space(3))) v4i128*)(l), 0, 0);
#else
  *(v8bf*)l = *(const v8bf*)g;
#endif
}

__device__ __forceinline__ void wait_async0() {
#ifdef HAVE_ASYNC_G2L
# if __has_builtin(__builtin_amdgcn_s_wait_asynccnt)
  __builtin_amdgcn_s_wait_asynccnt(0);
# else
  asm volatile("s_wait_asynccnt 0x0" ::: "memory");
# endif
#endif
}

__device__ __forceinline__ v16bf mk16(v8bf lo, v8bf hi) {
  v16bf r;
#pragma unroll
  for (int i = 0; i < 8; ++i) { r[i] = lo[i]; r[i + 8] = hi[i]; }
  return r;
}

// ---------------------------------------------------------------------------
// C[M,N] = A[M,K] (fp32->bf16) @ W[K,N] (fp32->bf16) + bias[N], fp32 acc.
// Block tile 128x128, BK=32, double-buffered LDS. 8 waves; wave tile 64x32.
// ---------------------------------------------------------------------------
__global__ __launch_bounds__(256) void gemm_bias_bf16(
    const float* __restrict__ A, const float* __restrict__ W,
    const float* __restrict__ bias, float* __restrict__ C,
    int M, int N, int K) {
  __shared__ __bf16 lA[2][128][40];   // [m][k], row stride 80B (16B aligned)
  __shared__ __bf16 lB[2][128][40];   // [n][k], N-major for contiguous B-frags

  const int t    = threadIdx.x;
  const int w    = t >> 5;
  const int lane = t & 31;
  const int lm   = lane & 15;
  const int hi   = lane >> 4;       // wave32 half
  const int ab   = hi * 8;          // A-frag K base (0 or 8)
  const int wm   = w >> 2;          // 0..1
  const int wn   = w & 3;           // 0..3
  const int mb   = blockIdx.y * 128;
  const int nb   = blockIdx.x * 128;

  auto loadRegs = [&](int kb, float4* ra, float4* rb) {
#pragma unroll
    for (int i = 0; i < 4; ++i) {
      int g = i * 256 + t;
      ra[i] = *(const float4*)(A + (size_t)(mb + (g >> 3)) * K + kb + (g & 7) * 4);
      rb[i] = *(const float4*)(W + (size_t)(kb + (g >> 5)) * N + nb + (g & 31) * 4);
    }
  };
  auto storeLds = [&](int buf, const float4* ra, const float4* rb) {
#pragma unroll
    for (int i = 0; i < 4; ++i) {
      int g = i * 256 + t;
      int m = g >> 3, f4 = g & 7;
      v4bf st;
      st[0] = (bf16)ra[i].x; st[1] = (bf16)ra[i].y;
      st[2] = (bf16)ra[i].z; st[3] = (bf16)ra[i].w;
      *(v4bf*)&lA[buf][m][f4 * 4] = st;
      int kk = g >> 5, nf4 = g & 31;
      lB[buf][nf4 * 4 + 0][kk] = (bf16)rb[i].x;
      lB[buf][nf4 * 4 + 1][kk] = (bf16)rb[i].y;
      lB[buf][nf4 * 4 + 2][kk] = (bf16)rb[i].z;
      lB[buf][nf4 * 4 + 3][kk] = (bf16)rb[i].w;
    }
  };

  v8f acc[4][2] = {};

  {
    float4 ra[4], rb[4];
    loadRegs(0, ra, rb);
    storeLds(0, ra, rb);
  }
  __syncthreads();

  for (int kb = 0; kb < K; kb += 32) {
    const int  cur  = (kb >> 5) & 1, nxt = cur ^ 1;
    const bool more = (kb + 32) < K;

    if (kb + 64 < K) {  // WGP-scope prefetch 2 tiles ahead (global_prefetch_b8)
      __builtin_prefetch(A + (size_t)(mb + (t >> 1)) * K + kb + 64 + (t & 1) * 16, 0, 3);
      __builtin_prefetch(W + (size_t)(kb + 64 + (t >> 3)) * N + nb + (t & 7) * 16, 0, 3);
    }

    float4 ra[4], rb[4];
    if (more) loadRegs(kb + 32, ra, rb);   // global loads in flight over WMMAs

    v16bf af[4];
#pragma unroll
    for (int mt = 0; mt < 4; ++mt) {
      const __bf16* r = &lA[cur][wm * 64 + mt * 16 + lm][0];
      af[mt] = mk16(*(const v8bf*)(r + ab), *(const v8bf*)(r + ab + 16));
    }
    v16bf bfm[2];
#pragma unroll
    for (int nt = 0; nt < 2; ++nt) {
      const __bf16* r = &lB[cur][wn * 32 + nt * 16 + lm][0];
      bfm[nt] = mk16(*(const v8bf*)(r + hi * 8), *(const v8bf*)(r + hi * 8 + 16));
    }
#pragma unroll
    for (int mt = 0; mt < 4; ++mt)
#pragma unroll
      for (int nt = 0; nt < 2; ++nt)
        acc[mt][nt] = WMMA_BF16(af[mt], bfm[nt], acc[mt][nt]);

    if (more) storeLds(nxt, ra, rb);
    __syncthreads();
  }

#pragma unroll
  for (int mt = 0; mt < 4; ++mt) {
#pragma unroll
    for (int nt = 0; nt < 2; ++nt) {
      int col  = nb + wn * 32 + nt * 16 + lm;
      float bv = bias[col];
#pragma unroll
      for (int v = 0; v < 8; ++v) {
        int row = mb + wm * 64 + mt * 16 + hi * 8 + v;
        C[(size_t)row * N + col] = acc[mt][nt][v] + bv;
      }
    }
  }
}

// ---------------------------------------------------------------------------
// RoPE on q,k + repack q,k,v to bf16 [B,H,S,D]. One thread per (b,s,h,pair).
// ---------------------------------------------------------------------------
__global__ __launch_bounds__(256) void rope_pack(
    const float* __restrict__ qkv, const float* __restrict__ cosT,
    const float* __restrict__ sinT, bf16* __restrict__ Qb,
    bf16* __restrict__ Kb, bf16* __restrict__ Vb) {
  int idx = blockIdx.x * 256 + threadIdx.x;   // B*S*H*64 = 2^22 exactly
  int p = idx & 63;
  int h = (idx >> 6) & 15;
  int s = (idx >> 10) & 2047;
  int b = idx >> 21;
  float c  = cosT[s * 64 + p];
  float sn = sinT[s * 64 + p];
  size_t base = ((size_t)(b * 2048 + s)) * 6144 + h * 128 + 2 * p;
  float qe = qkv[base],        qo = qkv[base + 1];
  float ke = qkv[base + 2048], ko = qkv[base + 2049];
  float ve = qkv[base + 4096], vo = qkv[base + 4097];
  size_t ob = ((size_t)((b * 16 + h) * 2048 + s)) * 128 + 2 * p;
  Qb[ob]     = (bf16)(qe * c - qo * sn);
  Qb[ob + 1] = (bf16)(qe * sn + qo * c);
  Kb[ob]     = (bf16)(ke * c - ko * sn);
  Kb[ob + 1] = (bf16)(ke * sn + ko * c);
  Vb[ob]     = (bf16)ve;
  Vb[ob + 1] = (bf16)vo;
}

// ---------------------------------------------------------------------------
// Causal flash attention. Grid (S/128, B*H), 256 threads = 8 waves.
// Each wave owns a 16-row q tile; waves share double-buffered K/V^T chunks.
// K chunks staged with GLOBAL_LOAD_ASYNC_TO_LDS_B128 (ASYNCcnt) when present.
// Out: fp32 [B*S, E] with cols h*128+d (ready for the proj GEMM).
// ---------------------------------------------------------------------------
__global__ __launch_bounds__(256) void attn_kernel(
    const bf16* __restrict__ Q, const bf16* __restrict__ K,
    const bf16* __restrict__ V, float* __restrict__ O) {
  __shared__ __bf16 sK[2][32][128];   // K chunk, row-major [k][d]
  __shared__ __bf16 sVt[2][128][32];  // V^T chunk [d][k]
  __shared__ __bf16 sP[8][16][32];    // per-wave probs staging

  const int t    = threadIdx.x;
  const int w    = t >> 5;
  const int lane = t & 31;
  const int lm   = lane & 15;
  const int hi   = lane >> 4;
  const int bh   = blockIdx.y;
  const int q0   = blockIdx.x * 128;
  const int qw   = q0 + w * 16;
  const size_t head = (size_t)bh * 2048 * 128;
  const float SCALE = 0.08838834764831845f;  // 1/sqrt(128)
  const int rrow = t >> 3, seg = t & 7;      // cooperative staging coords

  auto stageK = [&](int c, int buf) {
    const bf16* kp = K + head + (size_t)(c * 32 + rrow) * 128 + seg * 16;
    g2l_b128(kp,     &sK[buf][rrow][seg * 16]);
    g2l_b128(kp + 8, &sK[buf][rrow][seg * 16 + 8]);
  };
  auto loadV = [&](int c, v8bf& va, v8bf& vb) {
    const bf16* vp = V + head + (size_t)(c * 32 + rrow) * 128 + seg * 16;
    va = *(const v8bf*)(vp);
    vb = *(const v8bf*)(vp + 8);
  };
  auto scatterV = [&](int buf, v8bf va, v8bf vb) {
#pragma unroll
    for (int j = 0; j < 8; ++j) {
      sVt[buf][seg * 16 + j][rrow]     = va[j];
      sVt[buf][seg * 16 + 8 + j][rrow] = vb[j];
    }
  };

  // Q fragments (A-layout), loaded straight from global (row-major [s][d])
  v16bf qf[4];
  const bf16* qrow = Q + head + (size_t)(qw + lm) * 128;
#pragma unroll
  for (int dk = 0; dk < 4; ++dk)
    qf[dk] = mk16(*(const v8bf*)(qrow + dk * 32 + hi * 8),
                  *(const v8bf*)(qrow + dk * 32 + hi * 8 + 16));

  v8f o[8] = {};
  float mrow[8], lrow[8];
#pragma unroll
  for (int v = 0; v < 8; ++v) { mrow[v] = -3.0e38f; lrow[v] = 0.0f; }

  const int nChunks = (q0 + 128) >> 5;

  {  // prologue: stage chunk 0 into buffer 0
    stageK(0, 0);
    v8bf va, vb;
    loadV(0, va, vb);
    scatterV(0, va, vb);
  }
  wait_async0();
  __syncthreads();

  for (int c = 0; c < nChunks; ++c) {
    const int  cur  = c & 1, nxt = cur ^ 1;
    const bool more = (c + 1) < nChunks;
    const int  k0   = c * 32;

    v8bf va = {}, vb = {};
    if (more) {           // issue next chunk early: async K + V regs
      stageK(c + 1, nxt);
      loadV(c + 1, va, vb);
    }

    // S = Q @ K^T for this 16x32 chunk (two 16x16 score frags)
    v8f s0 = {}, s1 = {};
#pragma unroll
    for (int dk = 0; dk < 4; ++dk) {
      const bf16* r0 = &sK[cur][lm][dk * 32 + hi * 8];
      const bf16* r1 = &sK[cur][lm + 16][dk * 32 + hi * 8];
      v16bf b0 = mk16(*(const v8bf*)(r0), *(const v8bf*)(r0 + 16));
      v16bf b1 = mk16(*(const v8bf*)(r1), *(const v8bf*)(r1 + 16));
      s0 = WMMA_BF16(qf[dk], b0, s0);
      s1 = WMMA_BF16(qf[dk], b1, s1);
    }

    // online softmax (rows live in 16-lane halves; shfl width 16)
#pragma unroll
    for (int v = 0; v < 8; ++v) {
      int qg = qw + hi * 8 + v;
      float e0 = s0[v] * SCALE; if (k0 + lm > qg)      e0 = -3.0e38f;
      float e1 = s1[v] * SCALE; if (k0 + 16 + lm > qg) e1 = -3.0e38f;
      float r = fmaxf(e0, e1);
      r = fmaxf(r, __shfl_xor(r, 1, 16));
      r = fmaxf(r, __shfl_xor(r, 2, 16));
      r = fmaxf(r, __shfl_xor(r, 4, 16));
      r = fmaxf(r, __shfl_xor(r, 8, 16));
      float mnew = fmaxf(mrow[v], r);
      float corr = __expf(mrow[v] - mnew);
      float p0 = __expf(e0 - mnew);
      float p1 = __expf(e1 - mnew);
      float rs = p0 + p1;
      rs += __shfl_xor(rs, 1, 16);
      rs += __shfl_xor(rs, 2, 16);
      rs += __shfl_xor(rs, 4, 16);
      rs += __shfl_xor(rs, 8, 16);
      lrow[v] = lrow[v] * corr + rs;
      mrow[v] = mnew;
#pragma unroll
      for (int dn = 0; dn < 8; ++dn) o[dn][v] *= corr;
      sP[w][hi * 8 + v][lm]      = (bf16)p0;   // C-layout -> LDS
      sP[w][hi * 8 + v][lm + 16] = (bf16)p1;
    }
    // same-wave DS RAW: drain LDS stores before A-layout reload
    asm volatile("s_wait_dscnt 0x0" ::: "memory");

    const bf16* pr = &sP[w][lm][0];
    v16bf pa = mk16(*(const v8bf*)(pr + hi * 8), *(const v8bf*)(pr + hi * 8 + 16));
#pragma unroll
    for (int dn = 0; dn < 8; ++dn) {
      const bf16* vr = &sVt[cur][dn * 16 + lm][hi * 8];
      v16bf vv = mk16(*(const v8bf*)(vr), *(const v8bf*)(vr + 16));
      o[dn] = WMMA_BF16(pa, vv, o[dn]);
    }

    if (more) scatterV(nxt, va, vb);
    wait_async0();            // next chunk's async K loads complete
    __syncthreads();
  }

  const int b = bh >> 4, h = bh & 15;
#pragma unroll
  for (int v = 0; v < 8; ++v) {
    int srow = qw + hi * 8 + v;
    float inv = 1.0f / lrow[v];
    size_t rbase = ((size_t)(b * 2048 + srow)) * 2048 + h * 128;
#pragma unroll
    for (int dn = 0; dn < 8; ++dn)
      O[rbase + dn * 16 + lm] = o[dn][v] * inv;
  }
}

// ---------------------------------------------------------------------------
extern "C" void kernel_launch(void* const* d_in, const int* in_sizes, int n_in,
                              void* d_out, int out_size, void* d_ws, size_t ws_size,
                              hipStream_t stream) {
  const float* hidden   = (const float*)d_in[0];
  // d_in[1] = attention_mask (causal handled analytically)
  const float* rope_cos = (const float*)d_in[2];
  const float* rope_sin = (const float*)d_in[3];
  const float* c_attn_w = (const float*)d_in[4];
  const float* c_attn_b = (const float*)d_in[5];
  const float* c_proj_w = (const float*)d_in[6];
  const float* c_proj_b = (const float*)d_in[7];
  float* out = (float*)d_out;

  char* ws = (char*)d_ws;
  const size_t QKV_BYTES  = (size_t)4096 * 6144 * 4;         // 100,663,296
  const size_t HEAD_BYTES = (size_t)2 * 16 * 2048 * 128 * 2; // 16,777,216
  float* qkv  = (float*)ws;
  bf16*  Qb   = (bf16*)(ws + QKV_BYTES);
  bf16*  Kb   = (bf16*)(ws + QKV_BYTES + HEAD_BYTES);
  bf16*  Vb   = (bf16*)(ws + QKV_BYTES + 2 * HEAD_BYTES);
  float* attn = (float*)(ws + QKV_BYTES + 3 * HEAD_BYTES);

  gemm_bias_bf16<<<dim3(6144 / 128, 4096 / 128), 256, 0, stream>>>(
      hidden, c_attn_w, c_attn_b, qkv, 4096, 6144, 2048);
  rope_pack<<<dim3((2u * 2048 * 16 * 64) / 256), 256, 0, stream>>>(
      qkv, rope_cos, rope_sin, Qb, Kb, Vb);
  attn_kernel<<<dim3(2048 / 128, 2 * 16), 256, 0, stream>>>(Qb, Kb, Vb, attn);
  gemm_bias_bf16<<<dim3(2048 / 128, 4096 / 128), 256, 0, stream>>>(
      attn, c_proj_w, c_proj_b, out, 4096, 2048, 2048);
}
